// RetrievalHeadIndexDemultiplexing_34651796144513
// MI455X (gfx1250) — matrix operand order repaired and checked by reading, compiled, not run on hardware
//
#include <hip/hip_runtime.h>

// ---------------------------------------------------------------------------
// Problem constants (B=8, S=512, H=768, V=30522)
// ---------------------------------------------------------------------------
#define RB   8
#define SS   512
#define RR   4096          // B*S rows
#define HH   768
#define KK1  1536          // 2*H
#define VV   30522
#define NT2  1908          // ceil(V/16)

typedef __attribute__((ext_vector_type(16))) __bf16 v16bf;
typedef __attribute__((ext_vector_type(8)))  float  v8f;

// ---------------------------------------------------------------------------
// bf16 split helpers (round-to-nearest-even, pure bit ops: no bf16 VALU needed)
// ---------------------------------------------------------------------------
static __device__ __forceinline__ __bf16 f2bf(float f) {
    unsigned u = __builtin_bit_cast(unsigned, f);
    u += 0x7FFFu + ((u >> 16) & 1u);
    unsigned short h = (unsigned short)(u >> 16);
    return __builtin_bit_cast(__bf16, h);
}
static __device__ __forceinline__ float bf2f(__bf16 b) {
    unsigned short h = __builtin_bit_cast(unsigned short, b);
    return __builtin_bit_cast(float, (unsigned)h << 16);
}

// A-operand tile load for v_wmma_f32_16x16x32_bf16:
//   lane = half*16 + m;  VGPR0..3 hold K = half*8 .. half*8+7,
//                        VGPR4..7 hold K = 16 + half*8 .. +7   (two B128 loads)
static __device__ __forceinline__ v16bf load_a(const __bf16* p) {
    union { int4 q[2]; v16bf v; } u;
    u.q[0] = *(const int4*)(p);        // K chunk 0 (8 bf16 = 16B)
    u.q[1] = *(const int4*)(p + 16);   // K chunk 1
    return u.v;
}
// B-operand tile load: lane = half*16 + n; lane holds K = half*8 .. half*8+7
// and K = 16 + half*8 .. +7 (same order as load_a).
static __device__ __forceinline__ v16bf load_b(const __bf16* p) {
    union { int4 q[2]; v16bf v; } u;
    u.q[0] = *(const int4*)(p);
    u.q[1] = *(const int4*)(p + 16);
    return u.v;
}

// bf16x3 accumulation: c += Ahi*Bhi + Ahi*Blo + Alo*Bhi  (~16 mantissa bits)
static __device__ __forceinline__ v8f wmma3(v16bf ah, v16bf al, v16bf bh, v16bf bl, v8f c) {
    c = __builtin_amdgcn_wmma_f32_16x16x32_bf16(false, ah, false, bh, (short)0, c, false, false);
    c = __builtin_amdgcn_wmma_f32_16x16x32_bf16(false, ah, false, bl, (short)0, c, false, false);
    c = __builtin_amdgcn_wmma_f32_16x16x32_bf16(false, al, false, bh, (short)0, c, false, false);
    return c;
}

// ---------------------------------------------------------------------------
// Prep kernel: fp32 -> (bf16 hi, bf16 lo) split
// ---------------------------------------------------------------------------
__global__ void split_kernel(const float* __restrict__ src,
                             __bf16* __restrict__ hi, __bf16* __restrict__ lo, int n) {
    int i = blockIdx.x * blockDim.x + threadIdx.x;
    if (i >= n) return;
    float f = src[i];
    __bf16 h = f2bf(f);
    hi[i] = h;
    lo[i] = f2bf(f - bf2f(h));
}

// ---------------------------------------------------------------------------
// Prep kernel: build concat([gathered pos_embeds, features]) as split bf16
// ---------------------------------------------------------------------------
__global__ void build_a1_kernel(const float* __restrict__ features,
                                const int* __restrict__ labels,
                                __bf16* __restrict__ hi, __bf16* __restrict__ lo) {
    int i = blockIdx.x * blockDim.x + threadIdx.x;
    if (i >= RR * KK1) return;
    int row = i / KK1, k = i - row * KK1;
    int b = row >> 9, s = row & 511;
    float val;
    if (k < HH) {
        int lab = labels[(b << 9) + s];
        lab = (lab == -100) ? 0 : lab;
        val = features[((size_t)((b << 9) + lab)) * HH + k];
    } else {
        val = features[(size_t)row * HH + (k - HH)];
    }
    __bf16 h = f2bf(val);
    hi[i] = h;
    lo[i] = f2bf(val - bf2f(h));
}

// ---------------------------------------------------------------------------
// Fused GEMM1 (bf16x3 WMMA) + bias + exact GELU + LayerNorm + bf16-split of X.
// One block = one 16-row M-tile; 8 waves cover 48 N-tiles of 16; K = 1536.
// ---------------------------------------------------------------------------
#define G1_STRIDE 769   // padded LDS row stride (floats) to break bank patterns

__global__ __launch_bounds__(256) void gemm1_ln_kernel(
        const __bf16* __restrict__ Ah, const __bf16* __restrict__ Al,
        const __bf16* __restrict__ Wh, const __bf16* __restrict__ Wl,
        const float* __restrict__ dense_b,
        const float* __restrict__ ln_w, const float* __restrict__ ln_b,
        __bf16* __restrict__ Xh, __bf16* __restrict__ Xl) {
    __shared__ float gbuf[16 * G1_STRIDE];   // gelu(x) tile, 16 x 768
    __shared__ float redS[16][16];
    __shared__ float statMu[16];
    __shared__ float statRs[16];

    const int mbase = blockIdx.x * 16;
    const int lane  = threadIdx.x & 31;
    const int wave  = threadIdx.x >> 5;
    const int l16   = lane & 15;
    const int kh    = lane >> 4;            // which K-half this lane holds
    const int m     = mbase + l16;

    // ---- bf16x3 WMMA over all N-tiles owned by this wave ----
    for (int t = wave; t < HH / 16; t += 8) {
        const int n = t * 16 + l16;
        v8f c = {};
        for (int k = 0; k < KK1; k += 32) {
            v16bf ah = load_a(Ah + (size_t)m * KK1 + k + kh * 8);
            v16bf al = load_a(Al + (size_t)m * KK1 + k + kh * 8);
            v16bf bh = load_b(Wh + (size_t)n * KK1 + k + kh * 8);
            v16bf bl = load_b(Wl + (size_t)n * KK1 + k + kh * 8);
            c = wmma3(ah, al, bh, bl, c);
        }
        // D layout: lane holds column n, rows kh*8 + r. Apply bias + exact GELU.
        const float bias = dense_b[n];
        #pragma unroll
        for (int r = 0; r < 8; ++r) {
            float x = c[r] + bias;
            float g = 0.5f * x * (1.0f + erff(x * 0.70710678118654752f));
            gbuf[(kh * 8 + r) * G1_STRIDE + n] = g;
        }
    }
    __syncthreads();

    // ---- LayerNorm: 16 threads per row, 48 elements each (two-pass) ----
    const int row = threadIdx.x >> 4;
    const int c16 = threadIdx.x & 15;
    float s = 0.f;
    for (int j = 0; j < 48; ++j) s += gbuf[row * G1_STRIDE + c16 + 16 * j];
    redS[row][c16] = s;
    __syncthreads();
    if (c16 == 0) {
        float S = 0.f;
        for (int j = 0; j < 16; ++j) S += redS[row][j];
        statMu[row] = S * (1.0f / HH);
    }
    __syncthreads();
    const float mu = statMu[row];
    float q = 0.f;
    for (int j = 0; j < 48; ++j) {
        float d = gbuf[row * G1_STRIDE + c16 + 16 * j] - mu;
        q += d * d;
    }
    redS[row][c16] = q;
    __syncthreads();
    if (c16 == 0) {
        float Q = 0.f;
        for (int j = 0; j < 16; ++j) Q += redS[row][j];
        float var = Q * (1.0f / HH);
        statRs[row] = rsqrtf(fmaxf(var, 0.f) + 1e-12f);
    }
    __syncthreads();
    const float rs   = statRs[row];
    const int   grow = mbase + row;
    for (int j = 0; j < 48; ++j) {
        int col = c16 + 16 * j;
        float v = (gbuf[row * G1_STRIDE + col] - mu) * rs * ln_w[col] + ln_b[col];
        __bf16 h = f2bf(v);
        Xh[(size_t)grow * HH + col] = h;
        Xl[(size_t)grow * HH + col] = f2bf(v - bf2f(h));
    }
}

// ---------------------------------------------------------------------------
// GEMM2 (bf16x3 WMMA): logits = X (4096x768) * dec_w^T (768x30522) + dec_b.
// Grid: (239 N-groups of 8 wave-tiles) x (256 M-tiles). dec_w split (94 MB)
// lives in the 192 MB L2, so the only HBM stream is the 500 MB fp32 output.
// ---------------------------------------------------------------------------
__global__ __launch_bounds__(256) void gemm2_kernel(
        const __bf16* __restrict__ Xh, const __bf16* __restrict__ Xl,
        const __bf16* __restrict__ Wh, const __bf16* __restrict__ Wl,
        const float* __restrict__ dec_b, float* __restrict__ out) {
    const int lane  = threadIdx.x & 31;
    const int wave  = threadIdx.x >> 5;
    const int ntile = blockIdx.x * 8 + wave;
    const int nbase = ntile * 16;
    if (nbase >= VV) return;                 // wave-uniform: EXEC stays all-ones
    const int mbase = blockIdx.y * 16;
    const int l16   = lane & 15;
    const int kh    = lane >> 4;
    const int m     = mbase + l16;
    const int n     = nbase + l16;
    const int nc    = (n < VV) ? n : (VV - 1);   // clamp boundary loads (select, no branch)

    const __bf16* arh = Xh + (size_t)m  * HH;
    const __bf16* arl = Xl + (size_t)m  * HH;
    const __bf16* brh = Wh + (size_t)nc * HH;
    const __bf16* brl = Wl + (size_t)nc * HH;

    v8f c = {};
    for (int k = 0; k < HH; k += 32) {
        v16bf ah = load_a(arh + k + kh * 8);
        v16bf al = load_a(arl + k + kh * 8);
        v16bf bh = load_b(brh + k + kh * 8);
        v16bf bl = load_b(brl + k + kh * 8);
        c = wmma3(ah, al, bh, bl, c);
    }

    if (n < VV) {                            // per-lane masking only after WMMAs
        const float bias = dec_b[n];
        #pragma unroll
        for (int r = 0; r < 8; ++r) {
            int row = mbase + kh * 8 + r;
            out[(size_t)row * VV + n] = c[r] + bias;
        }
    }
}

// ---------------------------------------------------------------------------
// Launch
// ---------------------------------------------------------------------------
extern "C" void kernel_launch(void* const* d_in, const int* in_sizes, int n_in,
                              void* d_out, int out_size, void* d_ws, size_t ws_size,
                              hipStream_t stream) {
    const float* features = (const float*)d_in[0];
    const float* dense_w  = (const float*)d_in[1];
    const float* dense_b  = (const float*)d_in[2];
    const float* ln_w     = (const float*)d_in[3];
    const float* ln_b     = (const float*)d_in[4];
    const float* dec_w    = (const float*)d_in[5];
    const float* dec_b    = (const float*)d_in[6];
    const int*   labels   = (const int*)d_in[7];
    float* out = (float*)d_out;

    // Workspace carve-out (~136 MB total)
    char* ws = (char*)d_ws;
    size_t off = 0;
    auto carve = [&](size_t bytes) {
        char* p = ws + off;
        off = (off + bytes + 255) & ~(size_t)255;
        return p;
    };
    __bf16* A1h = (__bf16*)carve((size_t)RR * KK1 * 2);
    __bf16* A1l = (__bf16*)carve((size_t)RR * KK1 * 2);
    __bf16* W1h = (__bf16*)carve((size_t)HH * KK1 * 2);
    __bf16* W1l = (__bf16*)carve((size_t)HH * KK1 * 2);
    __bf16* Xh  = (__bf16*)carve((size_t)RR * HH * 2);
    __bf16* Xl  = (__bf16*)carve((size_t)RR * HH * 2);
    __bf16* W2h = (__bf16*)carve((size_t)VV * HH * 2);
    __bf16* W2l = (__bf16*)carve((size_t)VV * HH * 2);

    const int n_w1 = HH * KK1;
    const int n_w2 = VV * HH;
    const int n_a1 = RR * KK1;
    split_kernel<<<(n_w1 + 255) / 256, 256, 0, stream>>>(dense_w, W1h, W1l, n_w1);
    split_kernel<<<(n_w2 + 255) / 256, 256, 0, stream>>>(dec_w,   W2h, W2l, n_w2);
    build_a1_kernel<<<(n_a1 + 255) / 256, 256, 0, stream>>>(features, labels, A1h, A1l);

    gemm1_ln_kernel<<<RR / 16, 256, 0, stream>>>(A1h, A1l, W1h, W1l,
                                                 dense_b, ln_w, ln_b, Xh, Xl);

    dim3 g2((NT2 + 7) / 8, RR / 16);   // 239 x 256 blocks, 8 waves each
    gemm2_kernel<<<g2, 256, 0, stream>>>(Xh, Xl, W2h, W2l, dec_b, out);
}
